// EfficientTransformerBlock_69552700391614
// MI455X (gfx1250) — matrix operand order repaired and checked by reading, not compiled, not run on hardware
//
#include <hip/hip_runtime.h>
#include <hip/hip_bf16.h>
#include <math.h>

typedef _Float16 v16h __attribute__((ext_vector_type(16)));
typedef _Float16 v8h  __attribute__((ext_vector_type(8)));
typedef float    v8f  __attribute__((ext_vector_type(8)));

#define C_DIM   384
#define NHEAD   12
#define HDIM    32
#define T_TOK   49
#define NWIN    2048          // 8 * 16 * 16
#define M_ROWS  100352        // NWIN * T_TOK
#define N_QKV   1152
#define IMG_H   112
#define IMG_W   112

#if defined(__has_builtin)
#if __has_builtin(__builtin_amdgcn_global_load_async_to_lds_b128)
#define HAVE_ASYNC_LDS 1
#endif
#endif

#ifdef HAVE_ASYNC_LDS
typedef int v4i_gnu __attribute__((vector_size(16)));
typedef __attribute__((address_space(1))) v4i_gnu* gbl_v4i_p;
typedef __attribute__((address_space(3))) v4i_gnu* lds_v4i_p;
#endif

__device__ __forceinline__ v16h mk16(v8h lo, v8h hi) {
  v16h r;
#pragma unroll
  for (int i = 0; i < 8; ++i) { r[i] = lo[i]; r[i + 8] = hi[i]; }
  return r;
}

__device__ __forceinline__ float mishf(float v) {
  float sp = (v > 20.f) ? v : log1pf(expf(v));
  return v * tanhf(sp);
}

// -------- window partition: x[B,C,H,W] fp32 -> winX[M_ROWS, C] f16 ----------
__global__ __launch_bounds__(256) void pack_win_kernel(const float* __restrict__ x,
                                                       _Float16* __restrict__ winX) {
  size_t i = (size_t)blockIdx.x * 256 + threadIdx.x;
  if (i >= (size_t)M_ROWS * C_DIM) return;
  int c = (int)(i % C_DIM);
  size_t r = i / C_DIM;
  int t = (int)(r % T_TOK);
  int win = (int)(r / T_TOK);
  int ww = win & 15, hw = (win >> 4) & 15, b = win >> 8;
  int h = hw * 7 + t / 7, w = ww * 7 + t % 7;
  winX[i] = (_Float16)x[(((size_t)b * C_DIM + c) * IMG_H + h) * IMG_W + w];
}

// -------- weight transpose + f32 -> f16: src[N,K] -> dst[K,N] --------------
__global__ __launch_bounds__(256) void tconv_kernel(const float* __restrict__ src,
                                                    _Float16* __restrict__ dst,
                                                    int N, int K) {
  int i = blockIdx.x * 256 + threadIdx.x;
  if (i >= N * K) return;
  int n = i / K, k = i - n * K;
  dst[(size_t)k * N + n] = (_Float16)src[(size_t)n * K + k];
}

// -------- register-tiled f16 WMMA GEMM: out = A[M,K] @ B[K,N] + bias -------
// each wave computes a (16*MT) x (16*NT) macro-tile: MT*NT wmmas per k-step
// with only MT+NT fragment loads -> high wmma:vmem ratio, A/B panel reuse.
template <int MT, int NT>
__global__ __launch_bounds__(256) void gemm_wmma_tiled(
    const _Float16* __restrict__ A, const _Float16* __restrict__ B,
    const float* __restrict__ bias,
    _Float16* __restrict__ out16, float* __restrict__ out32,
    int M, int N, int K, int do_mish) {
  const int lane = threadIdx.x & 31;
  const int wave = blockIdx.x * 8 + (threadIdx.x >> 5);
  const int nblk = N / (16 * NT);
  const int total = (M / (16 * MT)) * nblk;
  if (wave >= total) return;                 // wave-uniform exit (EXEC all-1 for WMMA)
  const int mb = wave / nblk;
  const int nb = wave - mb * nblk;
  const int row0 = mb * (16 * MT), n0 = nb * (16 * NT);

  const int nlo = lane & 15;
  const int hi8 = (lane >> 4) << 3;
  const _Float16* ap = A + (size_t)(row0 + nlo) * K + hi8;  // A-frag: row nlo, K-half hi8
  const _Float16* bp = B + (size_t)hi8 * N + n0 + nlo;      // B-frag: col nlo, K-half hi8
  const size_t arow = (size_t)16 * K;
  const size_t bstep = (size_t)32 * N;

  v8f acc[MT][NT] = {};
  for (int k0 = 0; k0 < K; k0 += 32) {
    v16h af[MT], bf[NT];
#pragma unroll
    for (int i = 0; i < MT; ++i) {
      const _Float16* p = ap + (size_t)i * arow;
      af[i] = mk16(*(const v8h*)p, *(const v8h*)(p + 16));
    }
#pragma unroll
    for (int j = 0; j < NT; ++j) {
      const _Float16* p = bp + j * 16;
#pragma unroll
      for (int s = 0; s < 16; ++s) bf[j][s] = p[(size_t)(s + (s & 8)) * N];
    }
    ap += 32;
    bp += bstep;
#pragma unroll
    for (int i = 0; i < MT; ++i) __builtin_prefetch(ap + (size_t)i * arow, 0, 1);
    __builtin_prefetch(bp, 0, 1);            // global_prefetch_b8
#pragma unroll
    for (int i = 0; i < MT; ++i)
#pragma unroll
      for (int j = 0; j < NT; ++j)
        acc[i][j] = __builtin_amdgcn_wmma_f32_16x16x32_f16(false, af[i], false, bf[j],
                                                           (short)0, acc[i][j], false,
                                                           false);
  }

#pragma unroll
  for (int j = 0; j < NT; ++j) {
    const int n = n0 + j * 16 + nlo;
    const float bv = bias ? bias[n] : 0.f;
#pragma unroll
    for (int i = 0; i < MT; ++i) {
#pragma unroll
      for (int r = 0; r < 8; ++r) {
        const int m = row0 + i * 16 + r + hi8;  // D layout: VGPR r -> M=r (+8 upper half)
        float v = acc[i][j][r] + bv;
        if (do_mish) v = mishf(v);
        if (out32) out32[(size_t)m * N + n] = v;
        if (out16) out16[(size_t)m * N + n] = (_Float16)v;
      }
    }
  }
}

// -------- windowed attention: one wave per (window, head) ------------------
#define AT_WAVES 4
__global__ __launch_bounds__(128) void attn_kernel(const _Float16* __restrict__ qkv,
                                                   _Float16* __restrict__ oatt) {
  __shared__ _Float16 sKT[AT_WAVES][32 * 64];  // k^T, zero-padded keys
  __shared__ _Float16 sV[AT_WAVES][64 * 32];   // v, zero-padded keys
  __shared__ float    sS[AT_WAVES][16 * 64];   // scores for one m-tile
  __shared__ _Float16 sA[AT_WAVES][16 * 64];   // softmax(a) f16
  const int wv = threadIdx.x >> 5;
  const int lane = threadIdx.x & 31;
  const int task = blockIdx.x * AT_WAVES + wv; // grid covers exactly NWIN*NHEAD waves
  const int head = task % NHEAD;
  const int win = task / NHEAD;
  const _Float16* base = qkv + (size_t)win * T_TOK * N_QKV;
  const _Float16* qb = base + head * HDIM;
  const _Float16* kb = base + C_DIM + head * HDIM;
  const _Float16* vb = base + 2 * C_DIM + head * HDIM;
  _Float16* KT = sKT[wv];
  _Float16* Vl = sV[wv];
  float* Sm = sS[wv];
  _Float16* Am = sA[wv];

  // stage k^T [32 x 64] (transpose -> scalar ds stores, zero pad tokens 49..63)
  for (int idx = lane; idx < 2048; idx += 32) {
    int n = idx >> 5, kk = idx & 31;
    KT[kk * 64 + n] = (n < T_TOK) ? kb[(size_t)n * N_QKV + kk] : (_Float16)0.f;
  }
  // zero pad v rows 49..63
  for (int idx = lane; idx < (64 - T_TOK) * 32; idx += 32)
    Vl[T_TOK * 32 + idx] = (_Float16)0.f;
#ifdef HAVE_ASYNC_LDS
  // stage v [49 x 32] via async global->LDS DMA (16B chunks, ASYNCcnt-tracked)
  for (int idx = lane; idx < T_TOK * 4; idx += 32) {
    int kk = idx >> 2, c8 = (idx & 3) * 8;
    const _Float16* src = vb + (size_t)kk * N_QKV + c8;
    _Float16* dst = &Vl[kk * 32 + c8];
    __builtin_amdgcn_global_load_async_to_lds_b128(
        (gbl_v4i_p)(uintptr_t)src, (lds_v4i_p)(uintptr_t)dst, 0, 0);
  }
  asm volatile("s_wait_asynccnt 0x0" ::: "memory");
#else
  for (int idx = lane; idx < T_TOK * 32; idx += 32) {
    int kk = idx >> 5, d = idx & 31;
    Vl[kk * 32 + d] = vb[(size_t)kk * N_QKV + d];
  }
#endif
  asm volatile("s_wait_dscnt 0x0" ::: "memory");

  const float scale = 0.1767766952966369f;  // 1/sqrt(32)
  const int nlo = lane & 15;
  const int hi8 = (lane >> 4) << 3;

  for (int mt = 0; mt < 4; ++mt) {
    // q A-fragment (K = 32, single wmma step per n-tile)
    int mrow = mt * 16 + nlo;
    if (mrow >= T_TOK) mrow = T_TOK - 1;     // clamp pad rows (masked at store)
    const _Float16* qp = qb + (size_t)mrow * N_QKV + hi8;
    v16h qa = mk16(*(const v8h*)qp, *(const v8h*)(qp + 16));

    for (int nt = 0; nt < 4; ++nt) {
      v16h bf;
#pragma unroll
      for (int s = 0; s < 16; ++s) bf[s] = KT[(hi8 + s + (s & 8)) * 64 + nt * 16 + nlo];
      v8f sc = {};
      sc = __builtin_amdgcn_wmma_f32_16x16x32_f16(false, qa, false, bf, (short)0, sc,
                                                  false, false);
      int n = nt * 16 + nlo;
#pragma unroll
      for (int r = 0; r < 8; ++r) Sm[(r + hi8) * 64 + n] = sc[r] * scale;
    }
    asm volatile("s_wait_dscnt 0x0" ::: "memory");

    // softmax over the 49 valid keys; lane r < 16 owns local row r
    if (lane < 16) {
      float mx = -1e30f;
      for (int n = 0; n < T_TOK; ++n) mx = fmaxf(mx, Sm[lane * 64 + n]);
      float sum = 0.f;
      for (int n = 0; n < T_TOK; ++n) {
        float e = expf(Sm[lane * 64 + n] - mx);
        Sm[lane * 64 + n] = e;
        sum += e;
      }
      float inv = 1.f / sum;
      for (int n = 0; n < 64; ++n)
        Am[lane * 64 + n] = (n < T_TOK) ? (_Float16)(Sm[lane * 64 + n] * inv)
                                        : (_Float16)0.f;
    }
    asm volatile("s_wait_dscnt 0x0" ::: "memory");

    // o = a @ v : K = 64 (2 steps), N = 32 (2 n-tiles)
    v8f oc0 = {}, oc1 = {};
#pragma unroll
    for (int ks = 0; ks < 64; ks += 32) {
      int kb2 = ks + hi8;
      v16h af = mk16(*(const v8h*)&Am[nlo * 64 + kb2],
                     *(const v8h*)&Am[nlo * 64 + kb2 + 16]);
      v16h b0, b1;
#pragma unroll
      for (int s = 0; s < 16; ++s) {
        const int kr = (ks + hi8 + s + (s & 8)) * 32 + nlo;
        b0[s] = Vl[kr];
        b1[s] = Vl[kr + 16];
      }
      oc0 = __builtin_amdgcn_wmma_f32_16x16x32_f16(false, af, false, b0, (short)0, oc0,
                                                   false, false);
      oc1 = __builtin_amdgcn_wmma_f32_16x16x32_f16(false, af, false, b1, (short)0, oc1,
                                                   false, false);
    }
#pragma unroll
    for (int r = 0; r < 8; ++r) {
      int m = mt * 16 + r + hi8;
      if (m < T_TOK) {
        size_t o = ((size_t)win * T_TOK + m) * C_DIM + head * HDIM + nlo;
        oatt[o] = (_Float16)oc0[r];
        oatt[o + 16] = (_Float16)oc1[r];
      }
    }
  }
}

// -------- residual + LayerNorm over C=384, one row per block ---------------
__global__ __launch_bounds__(128) void ln_kernel(
    const float* __restrict__ in, const float* __restrict__ res_win,
    const float* __restrict__ res_bchw, const float* __restrict__ g,
    const float* __restrict__ be, float* __restrict__ y32,
    _Float16* __restrict__ y16, float* __restrict__ out_bchw) {
  const int r = blockIdx.x;
  const int tid = threadIdx.x;
  const int t = r % T_TOK;
  const int win = r / T_TOK;
  const int ww = win & 15, hw = (win >> 4) & 15, b = win >> 8;
  const int h = hw * 7 + t / 7, w = ww * 7 + t % 7;

  __shared__ float red[128];
  float v[3];
  float s = 0.f;
#pragma unroll
  for (int j = 0; j < 3; ++j) {
    int c = tid + j * 128;
    float xv = in[(size_t)r * C_DIM + c];
    if (res_win) xv += res_win[(size_t)r * C_DIM + c];
    if (res_bchw) xv += res_bchw[(((size_t)b * C_DIM + c) * IMG_H + h) * IMG_W + w];
    v[j] = xv;
    s += xv;
  }
  red[tid] = s;
  __syncthreads();
  for (int off = 64; off; off >>= 1) {
    if (tid < off) red[tid] += red[tid + off];
    __syncthreads();
  }
  float mean = red[0] * (1.f / C_DIM);
  __syncthreads();
  s = 0.f;
#pragma unroll
  for (int j = 0; j < 3; ++j) {
    float d = v[j] - mean;
    s += d * d;
  }
  red[tid] = s;
  __syncthreads();
  for (int off = 64; off; off >>= 1) {
    if (tid < off) red[tid] += red[tid + off];
    __syncthreads();
  }
  float inv = rsqrtf(red[0] * (1.f / C_DIM) + 1e-5f);
#pragma unroll
  for (int j = 0; j < 3; ++j) {
    int c = tid + j * 128;
    float o = (v[j] - mean) * inv * g[c] + be[c];
    if (y32) y32[(size_t)r * C_DIM + c] = o;
    if (y16) y16[(size_t)r * C_DIM + c] = (_Float16)o;
    if (out_bchw) out_bchw[(((size_t)b * C_DIM + c) * IMG_H + h) * IMG_W + w] = o;
  }
}

extern "C" void kernel_launch(void* const* d_in, const int* in_sizes, int n_in,
                              void* d_out, int out_size, void* d_ws, size_t ws_size,
                              hipStream_t stream) {
  (void)in_sizes; (void)n_in; (void)out_size; (void)ws_size;
  const float* x     = (const float*)d_in[0];
  const float* w_qkv = (const float*)d_in[1];
  const float* b_qkv = (const float*)d_in[2];
  const float* w_o   = (const float*)d_in[3];
  const float* b_o   = (const float*)d_in[4];
  const float* w_c1  = (const float*)d_in[5];
  const float* b_c1  = (const float*)d_in[6];
  const float* w_c2  = (const float*)d_in[7];
  const float* b_c2  = (const float*)d_in[8];
  const float* g1    = (const float*)d_in[9];
  const float* be1   = (const float*)d_in[10];
  const float* g2    = (const float*)d_in[11];
  const float* be2   = (const float*)d_in[12];

  char* ws = (char*)d_ws;
  auto al = [](size_t v) { return (v + 255) & ~(size_t)255; };
  const size_t M = M_ROWS;
  size_t off = 0;
  _Float16* winX = (_Float16*)(ws + off); off += al(M * C_DIM * 2);
  size_t offQ = off;
  _Float16* qkvB = (_Float16*)(ws + off); off += al(M * N_QKV * 2);
  float*    y32  = (float*)(ws + off);    off += al(M * C_DIM * 4);
  _Float16* y16  = (_Float16*)(ws + off); off += al(M * C_DIM * 2);
  _Float16* hB   = (_Float16*)(ws + off); off += al(M * 96 * 2);
  _Float16* wqkvT = (_Float16*)(ws + off); off += al((size_t)N_QKV * C_DIM * 2);
  _Float16* woT   = (_Float16*)(ws + off); off += al((size_t)C_DIM * C_DIM * 2);
  _Float16* wc1T  = (_Float16*)(ws + off); off += al((size_t)96 * C_DIM * 2);
  _Float16* wc2T  = (_Float16*)(ws + off); off += al((size_t)C_DIM * 96 * 2);
  // stream-ordered aliases (previous contents dead by the time these are written)
  _Float16* oatt = winX;              // after QKV GEMM, winX is dead
  float* proj  = (float*)(ws + offQ); // after attention, qkv buffer is dead
  float* c2out = (float*)(ws + offQ); // after LN1, proj is dead

  // weights -> [K,N] f16
  tconv_kernel<<<1728, 256, 0, stream>>>(w_qkv, wqkvT, N_QKV, C_DIM);
  tconv_kernel<<<576, 256, 0, stream>>>(w_o, woT, C_DIM, C_DIM);
  tconv_kernel<<<144, 256, 0, stream>>>(w_c1, wc1T, 96, C_DIM);
  tconv_kernel<<<144, 256, 0, stream>>>(w_c2, wc2T, C_DIM, 96);
  // window partition
  pack_win_kernel<<<150528, 256, 0, stream>>>(x, winX);

  // qkv = winX @ wqkvT + b_qkv   (M x 1152, K=384); 64x32 macro-tile per wave
  {
    int waves = (M_ROWS / 64) * (N_QKV / 32);
    gemm_wmma_tiled<4, 2><<<(waves + 7) / 8, 256, 0, stream>>>(
        winX, wqkvT, b_qkv, qkvB, nullptr, M_ROWS, N_QKV, C_DIM, 0);
  }
  // windowed multi-head attention
  attn_kernel<<<(NWIN * NHEAD) / AT_WAVES, 128, 0, stream>>>(qkvB, oatt);
  // out-proj (fp32 out); bias fused
  {
    int waves = (M_ROWS / 64) * (C_DIM / 32);
    gemm_wmma_tiled<4, 2><<<(waves + 7) / 8, 256, 0, stream>>>(
        oatt, woT, b_o, nullptr, proj, M_ROWS, C_DIM, C_DIM, 0);
  }
  // x = LN1(x + attn_out) -> y32 (residual source) + y16 (GEMM input)
  ln_kernel<<<M_ROWS, 128, 0, stream>>>(proj, nullptr, x, g1, be1, y32, y16, nullptr);
  // h = mish(y @ wc1T + b_c1)   (M x 96)
  {
    int waves = (M_ROWS / 64) * (96 / 32);
    gemm_wmma_tiled<4, 2><<<(waves + 7) / 8, 256, 0, stream>>>(
        y16, wc1T, b_c1, hB, nullptr, M_ROWS, 96, C_DIM, 1);
  }
  // c2 = h @ wc2T + b_c2        (M x 384, K=96)
  {
    int waves = (M_ROWS / 64) * (C_DIM / 32);
    gemm_wmma_tiled<4, 2><<<(waves + 7) / 8, 256, 0, stream>>>(
        hB, wc2T, b_c2, nullptr, c2out, M_ROWS, C_DIM, 96, 0);
  }
  // out = LN2(y + c2), scattered back to BCHW fp32
  ln_kernel<<<M_ROWS, 128, 0, stream>>>(c2out, y32, nullptr, g2, be2, nullptr, nullptr,
                                        (float*)d_out);
}
